// find_correspondence_14894946582630
// MI455X (gfx1250) — matrix operand order repaired and checked by reading, not compiled, not run on hardware
//
#include <hip/hip_runtime.h>

typedef __attribute__((ext_vector_type(16))) _Float16 v16h;
typedef __attribute__((ext_vector_type(8)))  float    v8f;

#define HWD 4096
// BETA * log2(e)
#define KB      72.13475204444817f
// log2(e) / (2*sigma^2) = 1.4426950408889634 / 50
#define GSCALE  0.028853900817779268f
#define STEP    (2.0f/63.0f)

__device__ __forceinline__ float fexp2(float x) {
#if __has_builtin(__builtin_amdgcn_exp2f)
  return __builtin_amdgcn_exp2f(x);
#else
  return exp2f(x);
#endif
}

__global__ __launch_bounds__(256) void corrflow_kernel(const float* __restrict__ corr,
                                                       float* __restrict__ out)
{
  __shared__ float T2[128];     // signed-index gaussian table: T2[64+d] = exp(-d^2/2s^2)
  __shared__ float rv[256];     // argmax partial values
  __shared__ int   ri[256];     // argmax partial indices
  __shared__ int   pmx[16];     // per-pixel argmax x
  __shared__ int   pmy[16];     // per-pixel argmax y
  __shared__ float pMb[16];     // per-pixel KB * max(c_max, 0)
  __shared__ float dacc[8*256]; // per-wave WMMA accumulators

  const int tid  = threadIdx.x;
  const int lane = tid & 31;
  const int wv   = tid >> 5;

  const int tile = blockIdx.x;        // 1024 tiles of 16 pixels
  const int p0   = tile << 4;         // first flattened pixel (b*4096 + i*64 + j)
  const int bI   = p0 >> 12;
  const int rem0 = p0 & 4095;         // pixel offset within the batch plane
  const float* base = corr + ((size_t)bI << 24);   // b * 4096*4096

  // ---- signed gaussian table (no abs needed on lookups) ----
  if (tid < 128) {
    float d = (float)(tid - 64);
    T2[tid] = fexp2(-d * d * GSCALE);
  }

  // ---- pass 1: per-pixel argmax over the 4096 source positions ----
  {
    const int pix = tid & 15;
    const int sg  = tid >> 4;                 // 16 threads cooperate per pixel
    const float* p = base + rem0 + pix + sg * HWD;
    float bv = -3.4e38f;
    int   bi = 0;
    int   s  = sg;
    #pragma unroll 4
    for (int k = 0; k < 256; ++k) {
      float c = *p;
      p += 16 * HWD;
      bool gt = (c > bv);
      bi = gt ? s : bi;
      bv = gt ? c : bv;
      s += 16;
    }
    rv[tid] = bv;
    ri[tid] = bi;
  }
  __syncthreads();
  if (tid < 16) {
    float bv = rv[tid];
    int   bi = ri[tid];
    #pragma unroll
    for (int t = 1; t < 16; ++t) {
      float v  = rv[tid + (t << 4)];
      int   i2 = ri[tid + (t << 4)];
      if (v > bv || (v == bv && i2 < bi)) { bv = v; bi = i2; }
    }
    pmx[tid] = bi & 63;
    pmy[tid] = bi >> 6;
    pMb[tid] = KB * fmaxf(bv, 0.0f);   // exact, overflow-safe softmax shift
  }
  __syncthreads();

  // ---- pass 2: softmax weights -> WMMA contraction with [1, xn, yn] ----
  const int pix   = lane & 15;     // A-matrix M = pixel (lanes 0-15 / 16-31)
  const int khalf = lane >> 4;     // A-matrix K half selector
  const int n     = lane & 15;     // B-matrix column held by this lane
  const int mx    = pmx[pix];
  const int my    = pmy[pix];
  const float Mb  = pMb[pix];
  const float kb16f  = (float)(khalf << 3);     // B K-range base (0 or 8)
  const float fsel1x = (n == 1) ? STEP : 0.0f;  // x0 contribution only for xn column
  const float fsel2  = (n == 2) ? 1.0f : 0.0f;  // yn column selector

  // chunk-invariant part of B: col0 = 1, col1 = xn ramp for x0=0, others 0
  v16h bstatic;
  #pragma unroll
  for (int e = 0; e < 16; ++e) {
    float v = (n == 0) ? 1.0f
            : (n == 1) ? fmaf(kb16f + (float)(e + ((e & 8) ? 8 : 0)), STEP, -1.0f)
            : 0.0f;
    bstatic[e] = (_Float16)v;
  }

  v8f acc = {};
  for (int it = 0; it < 16; ++it) {
    const int sb = (wv * 16 + it) * 32;   // 32 source positions per chunk
    const int y  = sb >> 6;               // constant row within chunk
    const int x0 = sb & 63;

    const float gyKb = T2[64 + y - my] * KB;
    const int   d0   = x0 + (khalf << 3) - mx;
    const float* tp  = &T2[64 + d0];      // per-chunk LDS base, imm offsets below
    const float* lp  = base + (sb + (khalf << 3)) * HWD + rem0 + pix;

    // A: 16 f16 softmax weights (16-bit A layout: e<8 -> K=khalf*8+e, e>=8 -> 16+khalf*8+(e-8))
    v16h a;
    #pragma unroll
    for (int e = 0; e < 16; ++e) {
      const int ofs = e + ((e & 8) ? 8 : 0);
      float c  = lp[ofs * HWD];
      float gx = tp[ofs];
      float t  = fexp2(fmaf(gx * gyKb, c, -Mb));
      a[e] = (_Float16)t;
    }

    // B: bstatic + per-chunk scalar (x0 shift for xn column, yn broadcast)
    const float ynv  = fmaf((float)y, STEP, -1.0f);
    const float addf = fmaf(fsel1x, (float)x0, fsel2 * ynv);
    const _Float16 ah = (_Float16)addf;
    v16h b;
    #pragma unroll
    for (int e = 0; e < 16; ++e)
      b[e] = (_Float16)(bstatic[e] + ah);

    acc = __builtin_amdgcn_wmma_f32_16x16x32_f16(
        /*neg_a=*/false, a, /*neg_b=*/false, b,
        /*c_mod=*/(short)0, acc, /*reuse_a=*/false, /*reuse_b=*/false);
  }

  // ---- reduce per-wave partial D matrices, emit grid + flow ----
  #pragma unroll
  for (int r = 0; r < 8; ++r)
    dacc[wv * 256 + lane * 8 + r] = acc[r];
  __syncthreads();

  if (tid < 16) {
    const int m  = tid;
    const int l0 = (m < 8) ? 0 : 16;   // lane group holding rows m / m+8
    const int r  = m & 7;              // D VGPR index
    float S = 0.0f, Sx = 0.0f, Sy = 0.0f;
    #pragma unroll
    for (int w = 0; w < 8; ++w) {
      const int o = w * 256 + r;
      S  += dacc[o + (l0 + 0) * 8];
      Sx += dacc[o + (l0 + 1) * 8];
      Sy += dacc[o + (l0 + 2) * 8];
    }
    const float gx = Sx / S;
    const float gy = Sy / S;

    const int rem = rem0 + m;
    const int i   = rem >> 6;
    const int j   = rem & 63;
    const float X = fmaf((float)j, STEP, -1.0f);
    const float Y = fmaf((float)i, STEP, -1.0f);

    // grid: (b,h,w,2) flat
    const int gbase = ((bI << 12) + rem) * 2;
    out[gbase + 0] = gx;
    out[gbase + 1] = gy;
    // flow: (b,2,h,w) flat, after 32768 grid floats
    const int fbase = 32768 + (bI << 13) + rem;
    out[fbase]        = gx - X;
    out[fbase + 4096] = gy - Y;
  }
}

extern "C" void kernel_launch(void* const* d_in, const int* in_sizes, int n_in,
                              void* d_out, int out_size, void* d_ws, size_t ws_size,
                              hipStream_t stream) {
  (void)in_sizes; (void)n_in; (void)out_size; (void)d_ws; (void)ws_size;
  const float* corr = (const float*)d_in[0];
  float* out = (float*)d_out;
  // 16384 target pixels / 16 per tile = 1024 workgroups of 8 waves
  corrflow_kernel<<<dim3(1024), dim3(256), 0, stream>>>(corr, out);
}
